// EdgeTamVideoRoPECrossAttention_11991548690518
// MI455X (gfx1250) — matrix operand order repaired and checked by reading, not compiled, not run on hardware
//
#include <hip/hip_runtime.h>
#include <hip/hip_bf16.h>

typedef __attribute__((ext_vector_type(16))) __bf16 v16bf;
typedef __attribute__((ext_vector_type(8)))  __bf16 v8bf;
typedef __attribute__((ext_vector_type(8)))  float  v8f;

#define HID   256
#define KVIN  64
#define NB    8
#define SQ    4096
#define SK    2304
#define MQ    (NB*SQ)   /* 32768 */
#define MK    (NB*SK)   /* 18432 */
#define NROPE 2240
#define GROUP 320
#define TEMPO 64
#define ASCALE 0.0625f  /* 1/sqrt(256) */

__device__ __forceinline__ v8f wmma_bf16(v16bf a, v16bf b, v8f c) {
  return __builtin_amdgcn_wmma_f32_16x16x32_bf16(false, a, false, b, (short)0, c,
                                                 false, false);
}
__device__ __forceinline__ v8f zero8() {
  v8f r;
#pragma unroll
  for (int i = 0; i < 8; ++i) r[i] = 0.f;
  return r;
}
// A fragment 16x32 bf16 (row-major source, row given by lane&15).
// lanes 0-15: K 0-7 & 16-23 ; lanes 16-31: K 8-15 & 24-31
__device__ __forceinline__ v16bf load_frag_a(const __bf16* row, bool hi) {
  const __bf16* p = row + (hi ? 8 : 0);
  v8bf x0 = *(const v8bf*)p;
  v8bf x1 = *(const v8bf*)(p + 16);
  v16bf r;
#pragma unroll
  for (int i = 0; i < 8; ++i) { r[i] = x0[i]; r[i + 8] = x1[i]; }
  return r;
}
// B fragment 32x16 bf16: lane holds column; source stored [col][k] row-major.
// lanes 0-15: K 0-7 & 16-23 ; lanes 16-31: K 8-15 & 24-31
__device__ __forceinline__ v16bf load_frag_b(const __bf16* colrow, bool hi) {
  const __bf16* p = colrow + (hi ? 8 : 0);
  v8bf x0 = *(const v8bf*)p;
  v8bf x1 = *(const v8bf*)(p + 16);
  v16bf r;
#pragma unroll
  for (int i = 0; i < 8; ++i) { r[i] = x0[i]; r[i + 8] = x1[i]; }
  return r;
}
// Async copy 64B: global (ga) -> LDS (lds byte address). offset: applies to both sides.
__device__ __forceinline__ void async_copy64(unsigned lds, unsigned long long ga) {
  asm volatile("global_load_async_to_lds_b128 %0, %1, off"
               :: "v"(lds), "v"(ga) : "memory");
  asm volatile("global_load_async_to_lds_b128 %0, %1, off offset:16"
               :: "v"(lds), "v"(ga) : "memory");
  asm volatile("global_load_async_to_lds_b128 %0, %1, off offset:32"
               :: "v"(lds), "v"(ga) : "memory");
  asm volatile("global_load_async_to_lds_b128 %0, %1, off offset:48"
               :: "v"(lds), "v"(ga) : "memory");
}
__device__ __forceinline__ unsigned lds_addr(const void* p) {
  return (unsigned)(unsigned long long)(uintptr_t)p;
}

__global__ void cvt_bf16_kernel(const float* __restrict__ in,
                                __bf16* __restrict__ out, long n) {
  long i = (long)blockIdx.x * blockDim.x + threadIdx.x;
  if (i < n) out[i] = (__bf16)in[i];
}

// C = A[M x 256] * W^T (W stored [N=256][K=256] bf16) + bias, then 2D RoPE, *1/16, bf16 out
__global__ __launch_bounds__(256) void qproj_rope_kernel(
    const __bf16* __restrict__ A, const __bf16* __restrict__ W,
    const float* __restrict__ bias, const float* __restrict__ cosT,
    const float* __restrict__ sinT, __bf16* __restrict__ out) {
  int tid = threadIdx.x, w = tid >> 5, lane = tid & 31;
  bool hi = lane >= 16;
  int col = lane & 15;
  int mbase = blockIdx.x * 16;
  int nbase = w * 32;
  int arow = mbase + col;
  v8f c0 = zero8(), c1 = zero8();
#pragma unroll
  for (int k0 = 0; k0 < HID; k0 += 32) {
    v16bf a  = load_frag_a(&A[(long)arow * HID + k0], hi);
    v16bf b0 = load_frag_b(&W[(long)(nbase + col) * HID + k0], hi);
    v16bf b1 = load_frag_b(&W[(long)(nbase + 16 + col) * HID + k0], hi);
    c0 = wmma_bf16(a, b0, c0);
    c1 = wmma_bf16(a, b1, c1);
  }
#pragma unroll
  for (int j = 0; j < 8; ++j) {
    int rowm = mbase + j + (hi ? 8 : 0);
    int pos = rowm & (SQ - 1);
#pragma unroll
    for (int t = 0; t < 2; ++t) {
      int n = nbase + t * 16 + col;
      float val = (t ? c1[j] : c0[j]) + bias[n];
      float other = __shfl_xor(val, 1, 32);
      float cs = cosT[(long)pos * HID + n];
      float sn = sinT[(long)pos * HID + n];
      float rot = (n & 1) ? other : -other;
      val = val * cs + rot * sn;
      out[(long)rowm * HID + n] = (__bf16)(val * ASCALE);
    }
  }
}

// C = A[M x 64] * W^T (W stored [256][64] bf16) + bias, optional selective RoPE.
// vt_mode=0: out[row][n] row-major. vt_mode=1: out stored per-batch transposed [b][n][s].
__global__ __launch_bounds__(256) void kvproj_kernel(
    const __bf16* __restrict__ A, const __bf16* __restrict__ W,
    const float* __restrict__ bias, const float* __restrict__ cosT,
    const float* __restrict__ sinT, __bf16* __restrict__ out, int do_rope,
    int vt_mode) {
  int tid = threadIdx.x, w = tid >> 5, lane = tid & 31;
  bool hi = lane >= 16;
  int col = lane & 15;
  int mbase = blockIdx.x * 16;
  int nbase = w * 32;
  int arow = mbase + col;
  v8f c0 = zero8(), c1 = zero8();
#pragma unroll
  for (int k0 = 0; k0 < KVIN; k0 += 32) {
    v16bf a  = load_frag_a(&A[(long)arow * KVIN + k0], hi);
    v16bf b0 = load_frag_b(&W[(long)(nbase + col) * KVIN + k0], hi);
    v16bf b1 = load_frag_b(&W[(long)(nbase + 16 + col) * KVIN + k0], hi);
    c0 = wmma_bf16(a, b0, c0);
    c1 = wmma_bf16(a, b1, c1);
  }
#pragma unroll
  for (int j = 0; j < 8; ++j) {
    int rowm = mbase + j + (hi ? 8 : 0);
    int bidx = rowm / SK;
    int s = rowm - bidx * SK;
    int pos = -1;
    if (do_rope && s < NROPE) {
      int tt = s % GROUP;
      if (tt >= TEMPO) pos = tt - TEMPO;
    }
#pragma unroll
    for (int t = 0; t < 2; ++t) {
      int n = nbase + t * 16 + col;
      float val = (t ? c1[j] : c0[j]) + bias[n];
      float other = __shfl_xor(val, 1, 32);
      if (pos >= 0) {
        float cs = cosT[(long)pos * HID + n];
        float sn = sinT[(long)pos * HID + n];
        float rot = (n & 1) ? other : -other;
        val = val * cs + rot * sn;
      }
      if (vt_mode)
        out[((long)bidx * HID + n) * SK + s] = (__bf16)val;
      else
        out[(long)rowm * HID + n] = (__bf16)val;
    }
  }
}

// Flash attention: one block = 128 query rows (8 waves x 16), loop over 72 key tiles of 32.
// K is [b][key][d] row-major; V is pre-transposed [b][d][key].
__global__ __launch_bounds__(256) void attn_kernel(
    const __bf16* __restrict__ Q, const __bf16* __restrict__ K,
    const __bf16* __restrict__ VtG, __bf16* __restrict__ Oout) {
  __shared__ __align__(16) __bf16 Kt[32 * HID];   // [key][d]
  __shared__ __align__(16) __bf16 Vt[HID * 32];   // [d][key]
  __shared__ __align__(16) __bf16 Pl[8][16 * 32]; // per-wave P scratch
  int tid = threadIdx.x, w = tid >> 5, lane = tid & 31;
  bool hi = lane >= 16;
  int col = lane & 15;
  int b  = blockIdx.x >> 5;
  int qb = blockIdx.x & 31;
  int qrow0 = b * SQ + qb * 128 + w * 16;
  int arow = qrow0 + col;

  v16bf qa[8];
#pragma unroll
  for (int k = 0; k < 8; ++k)
    qa[k] = load_frag_a(&Q[(long)arow * HID + k * 32], hi);

  v8f O[16];
#pragma unroll
  for (int t = 0; t < 16; ++t) O[t] = zero8();
  float mrow[8], lrow[8];
#pragma unroll
  for (int j = 0; j < 8; ++j) { mrow[j] = -1e30f; lrow[j] = 0.f; }

  const __bf16* Kb = K + (long)b * SK * HID;
  const __bf16* Vb = VtG + (long)b * HID * SK;

  // Per-thread staging addresses (tile-invariant parts)
  int kkk = tid >> 3, kd0 = (tid & 7) * 32;         // K: thread -> 64B chunk
  unsigned kt_lds = lds_addr(&Kt[kkk * HID + kd0]);
  unsigned vt_lds = lds_addr(&Vt[tid * 32]);        // V^T: thread -> one d-row (64B)

  for (int kt = 0; kt < SK / 32; ++kt) {
    int key0 = kt * 32;
    __syncthreads();  // everyone done reading previous tile
    async_copy64(kt_lds, (unsigned long long)(uintptr_t)
                 &Kb[(long)(key0 + kkk) * HID + kd0]);
    async_copy64(vt_lds, (unsigned long long)(uintptr_t)
                 &Vb[(long)tid * SK + key0]);
    if (kt + 1 < SK / 32) {  // prefetch next tile into caches
      __builtin_prefetch(&Kb[(long)(key0 + 32 + kkk) * HID + kd0], 0, 1);
      __builtin_prefetch(&Vb[(long)tid * SK + key0 + 32], 0, 1);
    }
    asm volatile("s_wait_asynccnt 0" ::: "memory");
    __syncthreads();  // all waves' async copies landed

    // S = Q * K^T (pre-scaled Q); two 16-key column subtiles
    v8f s0 = zero8(), s1 = zero8();
#pragma unroll
    for (int k = 0; k < 8; ++k) {
      v16bf b0 = load_frag_b(&Kt[col * HID + k * 32], hi);
      v16bf b1 = load_frag_b(&Kt[(col + 16) * HID + k * 32], hi);
      s0 = wmma_bf16(qa[k], b0, s0);
      s1 = wmma_bf16(qa[k], b1, s1);
    }

    // online softmax per row (16-lane shuffle reductions)
    float alpha[8];
#pragma unroll
    for (int j = 0; j < 8; ++j) {
      float mx = fmaxf(s0[j], s1[j]);
#pragma unroll
      for (int off = 8; off; off >>= 1) mx = fmaxf(mx, __shfl_xor(mx, off, 32));
      float mnew = fmaxf(mrow[j], mx);
      alpha[j] = __expf(mrow[j] - mnew);
      mrow[j] = mnew;
      float p0 = __expf(s0[j] - mnew);
      float p1 = __expf(s1[j] - mnew);
      s0[j] = p0; s1[j] = p1;
      float rs = p0 + p1;
#pragma unroll
      for (int off = 8; off; off >>= 1) rs += __shfl_xor(rs, off, 32);
      lrow[j] = lrow[j] * alpha[j] + rs;
    }

    // P: C-layout -> A-layout via per-wave LDS
#pragma unroll
    for (int j = 0; j < 8; ++j) {
      int r = j + (hi ? 8 : 0);
      Pl[w][r * 32 + col]      = (__bf16)s0[j];
      Pl[w][r * 32 + col + 16] = (__bf16)s1[j];
    }
    asm volatile("s_wait_dscnt 0" ::: "memory");
    v16bf pa = load_frag_a(&Pl[w][col * 32], hi);

    // O = O*alpha + P*V
#pragma unroll
    for (int t = 0; t < 16; ++t) {
#pragma unroll
      for (int j = 0; j < 8; ++j) O[t][j] *= alpha[j];
      v16bf bv = load_frag_b(&Vt[(t * 16 + col) * 32], hi);
      O[t] = wmma_bf16(pa, bv, O[t]);
    }
  }

#pragma unroll
  for (int t = 0; t < 16; ++t) {
#pragma unroll
    for (int j = 0; j < 8; ++j) {
      int r = qrow0 + j + (hi ? 8 : 0);
      int d = t * 16 + col;
      Oout[(long)r * HID + d] = (__bf16)(O[t][j] / lrow[j]);
    }
  }
}

// out(f32) = A[M x 256] * W^T (W [256][256] bf16) + bias
__global__ __launch_bounds__(256) void oproj_kernel(
    const __bf16* __restrict__ A, const __bf16* __restrict__ W,
    const float* __restrict__ bias, float* __restrict__ out) {
  int tid = threadIdx.x, w = tid >> 5, lane = tid & 31;
  bool hi = lane >= 16;
  int col = lane & 15;
  int mbase = blockIdx.x * 16;
  int nbase = w * 32;
  int arow = mbase + col;
  v8f c0 = zero8(), c1 = zero8();
#pragma unroll
  for (int k0 = 0; k0 < HID; k0 += 32) {
    v16bf a  = load_frag_a(&A[(long)arow * HID + k0], hi);
    v16bf b0 = load_frag_b(&W[(long)(nbase + col) * HID + k0], hi);
    v16bf b1 = load_frag_b(&W[(long)(nbase + 16 + col) * HID + k0], hi);
    c0 = wmma_bf16(a, b0, c0);
    c1 = wmma_bf16(a, b1, c1);
  }
#pragma unroll
  for (int j = 0; j < 8; ++j) {
    int rowm = mbase + j + (hi ? 8 : 0);
#pragma unroll
    for (int t = 0; t < 2; ++t) {
      int n = nbase + t * 16 + col;
      out[(long)rowm * HID + n] = (t ? c1[j] : c0[j]) + bias[n];
    }
  }
}

extern "C" void kernel_launch(void* const* d_in, const int* in_sizes, int n_in,
                              void* d_out, int out_size, void* d_ws, size_t ws_size,
                              hipStream_t stream) {
  (void)in_sizes; (void)n_in; (void)out_size; (void)ws_size;
  const float* query = (const float*)d_in[0];
  const float* key_  = (const float*)d_in[1];
  const float* value = (const float*)d_in[2];
  const float* q_w = (const float*)d_in[3];
  const float* q_b = (const float*)d_in[4];
  const float* k_w = (const float*)d_in[5];
  const float* k_b = (const float*)d_in[6];
  const float* v_w = (const float*)d_in[7];
  const float* v_b = (const float*)d_in[8];
  const float* o_w = (const float*)d_in[9];
  const float* o_b = (const float*)d_in[10];
  const float* cosq = (const float*)d_in[11];
  const float* sinq = (const float*)d_in[12];
  const float* cosk = (const float*)d_in[13];
  const float* sink = (const float*)d_in[14];

  char* p = (char*)d_ws;
  auto alloc = [&](size_t bytes) {
    char* r = p;
    p += (bytes + 255) & ~(size_t)255;
    return r;
  };
  __bf16* qin   = (__bf16*)alloc((size_t)MQ * HID * 2);
  __bf16* kin   = (__bf16*)alloc((size_t)MK * KVIN * 2);
  __bf16* vin   = (__bf16*)alloc((size_t)MK * KVIN * 2);
  __bf16* qwb   = (__bf16*)alloc((size_t)HID * HID * 2);
  __bf16* kwb   = (__bf16*)alloc((size_t)HID * KVIN * 2);
  __bf16* vwb   = (__bf16*)alloc((size_t)HID * KVIN * 2);
  __bf16* owb   = (__bf16*)alloc((size_t)HID * HID * 2);
  __bf16* qpost = (__bf16*)alloc((size_t)MQ * HID * 2);
  __bf16* kpost = (__bf16*)alloc((size_t)MK * HID * 2);
  __bf16* vpostT= (__bf16*)alloc((size_t)MK * HID * 2);  // [b][d][key]
  __bf16* aout  = (__bf16*)alloc((size_t)MQ * HID * 2);

  auto cvt = [&](const float* in, __bf16* out, long n) {
    cvt_bf16_kernel<<<(unsigned)((n + 255) / 256), 256, 0, stream>>>(in, out, n);
  };
  cvt(query, qin, (long)MQ * HID);
  cvt(key_, kin, (long)MK * KVIN);
  cvt(value, vin, (long)MK * KVIN);
  cvt(q_w, qwb, (long)HID * HID);
  cvt(k_w, kwb, (long)HID * KVIN);
  cvt(v_w, vwb, (long)HID * KVIN);
  cvt(o_w, owb, (long)HID * HID);

  qproj_rope_kernel<<<MQ / 16, 256, 0, stream>>>(qin, qwb, q_b, cosq, sinq, qpost);
  kvproj_kernel<<<MK / 16, 256, 0, stream>>>(kin, kwb, k_b, cosk, sink, kpost, 1, 0);
  kvproj_kernel<<<MK / 16, 256, 0, stream>>>(vin, vwb, v_b, cosk, sink, vpostT, 0, 1);
  attn_kernel<<<NB * (SQ / 128), 256, 0, stream>>>(qpost, kpost, vpostT, aout);
  oproj_kernel<<<MQ / 16, 256, 0, stream>>>(aout, owb, o_b, (float*)d_out);
}
